// RAE_24455543783983
// MI455X (gfx1250) — matrix operand order repaired and checked by reading, not compiled, not run on hardware
//
#include <hip/hip_runtime.h>

typedef unsigned short u16;
typedef unsigned int   u32;
typedef __attribute__((ext_vector_type(16))) __bf16 v16bf;
typedef __attribute__((ext_vector_type(8)))  float   v8f;
typedef __attribute__((ext_vector_type(8)))  u32     v8u;

union FragU { v8u u; v16bf h; };

__device__ __forceinline__ u16 f2bf(float f) {
  u32 u = __float_as_uint(f);
  u += 0x7fffu + ((u >> 16) & 1u);   // round-to-nearest-even
  return (u16)(u >> 16);
}

// ---------------------------------------------------------------------------
// Pack W [512,1024] fp32 -> bf16 fragments in WMMA B-matrix lane order.
// Fragment (nt, kt): B tile 32(K) x 16(N). Lane L holds N = nt*16 + (L&15),
// K = kt*32 + (L>>4)*16 + j, j=0..15, 16 halfs contiguous per lane.
// ---------------------------------------------------------------------------
__global__ void rae_pack_w(const float* __restrict__ W, u32* __restrict__ packedW) {
  int t = blockIdx.x * blockDim.x + threadIdx.x;   // [0, 32768)
  int lane = t & 31;
  int kt   = (t >> 5) & 31;
  int nt   = t >> 10;
  int n  = nt * 16 + (lane & 15);
  int k0 = kt * 32 + (lane >> 4) * 16;
  const float* src = W + (long)n * 1024 + k0;
  u32* dst = packedW + (long)t * 8;
#pragma unroll
  for (int j = 0; j < 8; ++j) {
    u32 lo = f2bf(src[2 * j]);
    u32 hi = f2bf(src[2 * j + 1]);
    dst[j] = lo | (hi << 16);
  }
}

// ---------------------------------------------------------------------------
// Shared inner GEMM: one wave computes MT 16-row M-tiles x 4 N-tiles (64 cols)
// over K=1024. A in LDS, row layout = two 520-half sub-rows (stride 1040):
// half K lives at rowBase + K + 8*(K>=512). B from global packedW, immediate
// offsets. 4*MT WMMAs per 4 B fragments -> B latency amortized 4x (MT=4).
// ---------------------------------------------------------------------------
template<int MT>
__device__ __forceinline__ void gemm_strip(const u16* (&aBase)[MT],   // + rlo*1040 + hi*16 pre-applied
                                           const v8u* __restrict__ bPtr,
                                           v8f (&acc)[MT][4]) {
#pragma unroll
  for (int mt = 0; mt < MT; ++mt)
#pragma unroll
    for (int t = 0; t < 4; ++t) {
      v8f z = {0.f, 0.f, 0.f, 0.f, 0.f, 0.f, 0.f, 0.f};
      acc[mt][t] = z;
    }
#pragma unroll
  for (int kt = 0; kt < 32; ++kt) {
    const int c0 = kt * 32 + ((kt >= 16) ? 8 : 0);   // split-row pad skip
    FragU a[MT];
#pragma unroll
    for (int mt = 0; mt < MT; ++mt) {
      uint4 lo4 = *(const uint4*)(aBase[mt] + c0);
      uint4 hi4 = *(const uint4*)(aBase[mt] + c0 + 8);
      a[mt].u[0] = lo4.x; a[mt].u[1] = lo4.y; a[mt].u[2] = lo4.z; a[mt].u[3] = lo4.w;
      a[mt].u[4] = hi4.x; a[mt].u[5] = hi4.y; a[mt].u[6] = hi4.z; a[mt].u[7] = hi4.w;
    }
    FragU b[4];
#pragma unroll
    for (int t = 0; t < 4; ++t) b[t].u = bPtr[t * 1024 + kt * 32];
#pragma unroll
    for (int t = 0; t < 4; ++t)
#pragma unroll
      for (int mt = 0; mt < MT; ++mt)
        acc[mt][t] = __builtin_amdgcn_wmma_f32_16x16x32_bf16(
            false, a[mt].h, false, b[t].h, (short)0, acc[mt][t], false, false);
  }
}

// ---------------------------------------------------------------------------
// Big levels (d = 14..7, M multiple of 64): block = 64 rows x 512 cols,
// 8 waves each own 64 rows x 64 cols (MT=4). A tile (64 rows x 1024) staged
// once to LDS (133 KB, split-row layout); level 14 converts fp32 inp inline.
// ---------------------------------------------------------------------------
__global__ void __launch_bounds__(256) rae_level64(
    const void* __restrict__ src,          // child states (bf16) or inp (fp32)
    u16* __restrict__ statesOut,
    const v8u* __restrict__ packedW,
    const float* __restrict__ bias,
    int nodeStart, int childStart, int srcIsF32) {
  __shared__ __align__(16) u16 smem[64 * 1040];    // 133,120 B

  const int tid  = threadIdx.x;
  const int lane = tid & 31;
  const int wave = tid >> 5;
  const int rlo  = lane & 15;
  const int hi   = lane >> 4;
  const long mBase = (long)blockIdx.x * 64;

  // stage 64 A rows: 128 chunks(16B)/row, coalesced; pad +8 halfs at K=512
  for (int c = tid; c < 64 * 128; c += 256) {
    int r = c >> 7, j = c & 127;
    long rowElem = (childStart + 2 * (mBase + r)) * 512;
    uint4 val;
    if (srcIsF32) {
      const float4* s = (const float4*)((const float*)src + rowElem) + 2 * j;
      float4 f0 = s[0], f1 = s[1];
      val.x = (u32)f2bf(f0.x) | ((u32)f2bf(f0.y) << 16);
      val.y = (u32)f2bf(f0.z) | ((u32)f2bf(f0.w) << 16);
      val.z = (u32)f2bf(f1.x) | ((u32)f2bf(f1.y) << 16);
      val.w = (u32)f2bf(f1.z) | ((u32)f2bf(f1.w) << 16);
    } else {
      val = *((const uint4*)((const u16*)src + rowElem) + j);
    }
    *(uint4*)(smem + r * 1040 + j * 8 + ((j >= 64) ? 8 : 0)) = val;
  }
  __syncthreads();

  const u16* aBase[4];
#pragma unroll
  for (int mt = 0; mt < 4; ++mt)
    aBase[mt] = smem + (mt * 16 + rlo) * 1040 + hi * 16;
  const v8u* bPtr = packedW + (long)wave * 4 * 1024 + lane;

  v8f acc[4][4];
  gemm_strip<4>(aBase, bPtr, acc);

  // C/D layout: VGPR v, lane L -> row = v + (L>=16)*8, col = L%16
#pragma unroll
  for (int mt = 0; mt < 4; ++mt)
#pragma unroll
    for (int t = 0; t < 4; ++t) {
      const int col = wave * 64 + t * 16 + rlo;
      const float bb = bias[col];
      u16* outP = statesOut + (long)(nodeStart + mBase + mt * 16 + hi * 8) * 512 + col;
#pragma unroll
      for (int v = 0; v < 8; ++v)
        outP[(long)v * 512] = f2bf(tanhf(acc[mt][t][v] + bb));
    }
}

// ---------------------------------------------------------------------------
// Tail level helper: one level of M rows entirely in LDS (rows stride 520).
// ---------------------------------------------------------------------------
template<int MT>
__device__ __forceinline__ void tail_level(const u16* __restrict__ child,
                                           u16* __restrict__ outB,
                                           const v8u* __restrict__ bPtr,
                                           const float* __restrict__ bias,
                                           int M, int rlo, int hi, int wave,
                                           float* finalOut) {
  const u16* aBase[MT];
#pragma unroll
  for (int mt = 0; mt < MT; ++mt) {
    int r = mt * 16 + rlo;
    if (r > M - 1) r = M - 1;                 // clamp keeps EXEC full
    aBase[mt] = child + r * 1040 + hi * 16;   // child rows 2m,2m+1 at m*1040
  }
  v8f acc[MT][4];
  gemm_strip<MT>(aBase, bPtr, acc);

#pragma unroll
  for (int mt = 0; mt < MT; ++mt)
#pragma unroll
    for (int t = 0; t < 4; ++t) {
      const int col = wave * 64 + t * 16 + rlo;
      const float bb = bias[col];
#pragma unroll
      for (int v = 0; v < 8; ++v) {
        const int m = mt * 16 + hi * 8 + v;
        if (m < M) {
          float val = tanhf(acc[mt][t][v] + bb);
          outB[m * 520 + col] = f2bf(val);
          if (finalOut != nullptr && m == 0) finalOut[col] = val;
        }
      }
    }
}

// ---------------------------------------------------------------------------
// Fused tail (d = 6..0): one workgroup. Stage the 128 level-7 output rows
// (nodes 127..254) into LDS, then ping-pong levels through two LDS buffers
// with __syncthreads() as level barrier. Root written fp32 to d_out.
// ---------------------------------------------------------------------------
__global__ void __launch_bounds__(256) rae_tail(
    const u16* __restrict__ states,
    const v8u* __restrict__ packedW,
    const float* __restrict__ bias,
    float* __restrict__ out) {
  __shared__ __align__(16) u16 tsm[(128 + 64) * 520];   // 199,680 B
  u16* buf0 = tsm;                 // 128-row buffer
  u16* buf1 = tsm + 128 * 520;     // 64-row buffer

  const int tid  = threadIdx.x;
  const int lane = tid & 31;
  const int wave = tid >> 5;
  const int rlo  = lane & 15;
  const int hi   = lane >> 4;

  // stage 128 child state rows (512 halfs each) -> buf0, stride 520
  for (int c = tid; c < 128 * 64; c += 256) {
    int r = c >> 6, j = c & 63;
    uint4 val = *((const uint4*)(states + (long)(127 + r) * 512) + j);
    *(uint4*)(buf0 + r * 520 + j * 8) = val;
  }
  __syncthreads();

  const v8u* bPtr = packedW + (long)wave * 4 * 1024 + lane;

  int p = 0;
  for (int d = 6; d >= 0; --d) {
    const int M = 1 << d;
    const u16* child = (p == 0) ? buf0 : buf1;
    u16* outB = (p == 0) ? buf1 : buf0;
    if (M == 64)
      tail_level<4>(child, outB, bPtr, bias, M, rlo, hi, wave, nullptr);
    else if (M == 32)
      tail_level<2>(child, outB, bPtr, bias, M, rlo, hi, wave, nullptr);
    else
      tail_level<1>(child, outB, bPtr, bias, M, rlo, hi, wave,
                    (M == 1) ? out : nullptr);
    p ^= 1;
    __syncthreads();
  }
}

// ---------------------------------------------------------------------------
// Host: pack W, 8 big-level launches (d=14..7), 1 fused-tail launch.
// Scratch in d_ws: [0, 32767*512*2) bf16 states; [+, +1MB) packed W.
// ---------------------------------------------------------------------------
extern "C" void kernel_launch(void* const* d_in, const int* in_sizes, int n_in,
                              void* d_out, int out_size, void* d_ws, size_t ws_size,
                              hipStream_t stream) {
  (void)in_sizes; (void)n_in; (void)out_size; (void)ws_size;
  const float* inp = (const float*)d_in[3];
  const float* W   = (const float*)d_in[4];
  const float* b   = (const float*)d_in[5];

  u16* states = (u16*)d_ws;
  const size_t statesBytes = (size_t)32767 * 512 * sizeof(u16);  // 256B multiple
  u32* packedW = (u32*)((char*)d_ws + statesBytes);

  rae_pack_w<<<128, 256, 0, stream>>>(W, packedW);

  for (int d = 14; d >= 7; --d) {
    int M = 1 << d;                       // 16384 .. 128, multiples of 64
    const void* src = (d == 14) ? (const void*)inp : (const void*)states;
    rae_level64<<<M / 64, 256, 0, stream>>>(
        src, states, (const v8u*)packedW, b,
        (1 << d) - 1, (1 << (d + 1)) - 1, (d == 14) ? 1 : 0);
  }

  rae_tail<<<1, 256, 0, stream>>>(states, (const v8u*)packedW, b, (float*)d_out);
}
